// SignalEncoder_13924283974266
// MI455X (gfx1250) — matrix operand order repaired and checked by reading, not compiled, not run on hardware
//
#include <hip/hip_runtime.h>
#include <hip/hip_bf16.h>
#include <stdint.h>
#include <stddef.h>

// ---------------- problem constants (from reference) ----------------
#define T_LEN   1000000
#define CCH     64
#define KSZ     9
#define M_LEN   350000          // int(T * 0.35)
#define NORM_MEAN 0.35f
#define NB      ((T_LEN + 255) / 256)     // 3907 scan blocks
#define KTOT    (KSZ * CCH)               // 576 = im2col K
#define NKSTEP  (KTOT / 32)               // 18 WMMA K-steps
#define MT      (M_LEN / 16)              // 21875 output row tiles
#define LDSPAIR 33                        // padded LDS row stride in u32 pairs

typedef __attribute__((ext_vector_type(16))) __bf16    v16bf;
typedef __attribute__((ext_vector_type(8)))  float     v8f;
typedef __attribute__((ext_vector_type(8)))  uint32_t  v8u;

// ---------------- bf16 split helpers (fp32 = hi + lo) ----------------
__device__ __forceinline__ unsigned short f2bf(float f) {
  uint32_t u = __builtin_bit_cast(uint32_t, f);
  u += 0x7fffu + ((u >> 16) & 1u);        // round-to-nearest-even
  return (unsigned short)(u >> 16);
}
__device__ __forceinline__ float bf2f(unsigned short b) {
  return __builtin_bit_cast(float, (uint32_t)b << 16);
}
__device__ __forceinline__ uint32_t pack_split(float x) {   // bf16hi | bf16lo<<16
  unsigned short hb = f2bf(x);
  unsigned short lb = f2bf(x - bf2f(hb));
  return (uint32_t)hb | ((uint32_t)lb << 16);
}
__device__ __forceinline__ float sigmoidf_(float x) { return 1.f / (1.f + expf(-x)); }
__device__ __forceinline__ float siluf_(float x)    { return x / (1.f + expf(-x)); }

// ---------------- stage 1: predictor conv -> weights, moves + block sums ----
__global__ __launch_bounds__(256) void predictor_kernel(
    const float* __restrict__ x, const float* __restrict__ wp,
    const float* __restrict__ bp, float* __restrict__ weights,
    float* __restrict__ moves, float* __restrict__ bsum) {
  __shared__ float red[256];
  int tid = threadIdx.x;
  int t = blockIdx.x * 256 + tid;
  float mv = 0.f;
  if (t < T_LEN) {
    float s0 = bp[0], s1 = bp[1];
#pragma unroll
    for (int k = 0; k < KSZ; ++k) {
      int u = t + k - KSZ / 2;
      float xv = (u >= 0 && u < T_LEN) ? x[u] : 0.f;
      float x2 = xv * xv;
      // wp shape (2,2,9): [o][i][k]
      s0 += wp[k] * xv + wp[KSZ + k] * x2;
      s1 += wp[2 * KSZ + k] * xv + wp[3 * KSZ + k] * x2;
    }
    weights[t] = sigmoidf_(s0);
    mv = sigmoidf_(s1) * NORM_MEAN;
    moves[t] = mv;
  }
  red[tid] = mv;
  __syncthreads();
  for (int s = 128; s > 0; s >>= 1) {
    if (tid < s) red[tid] += red[tid + s];
    __syncthreads();
  }
  if (tid == 0) bsum[blockIdx.x] = red[0];
}

// ---------------- exclusive scan of the block sums (single block) -----------
__global__ __launch_bounds__(256) void scan_bsums_kernel(float* __restrict__ bs, int n) {
  __shared__ float buf[256];
  int tid = threadIdx.x;
  float carry = 0.f;                       // identical across threads
  for (int base = 0; base < n; base += 256) {
    int i = base + tid;
    float v = (i < n) ? bs[i] : 0.f;
    buf[tid] = v;
    __syncthreads();
    for (int off = 1; off < 256; off <<= 1) {
      float a = (tid >= off) ? buf[tid - off] : 0.f;
      __syncthreads();
      buf[tid] += a;
      __syncthreads();
    }
    if (i < n) bs[i] = carry + buf[tid] - v;   // exclusive prefix
    float total = buf[255];
    __syncthreads();
    carry += total;
  }
}

// ---------------- pos = cumsum(moves), in place --------------------------
__global__ __launch_bounds__(256) void pos_scan_kernel(
    float* __restrict__ mv_pos, const float* __restrict__ bsum) {
  __shared__ float buf[256];
  int tid = threadIdx.x;
  int t = blockIdx.x * 256 + tid;
  float v = (t < T_LEN) ? mv_pos[t] : 0.f;
  buf[tid] = v;
  __syncthreads();
  for (int off = 1; off < 256; off <<= 1) {
    float a = (tid >= off) ? buf[tid - off] : 0.f;
    __syncthreads();
    buf[tid] += a;
    __syncthreads();
  }
  if (t < T_LEN) mv_pos[t] = bsum[blockIdx.x] + buf[tid];   // inclusive cumsum
}

// ---------------- pack conv weights into WMMA B-fragment pair planes -----
// wH/wL[((tile*18 + kk)*32 + lane)*8 + v] = bf16(B[K=2v]) | bf16(B[K=2v+1])<<16
// where lane: n = lane&15 (col), half = lane>>4; K = kk*32 + 2v + half*16.
__global__ __launch_bounds__(256) void pack_weights_kernel(
    const float* __restrict__ w, uint32_t* __restrict__ wH, uint32_t* __restrict__ wL) {
  int idx = blockIdx.x * 256 + threadIdx.x;
  if (idx >= 4 * NKSTEP * 32 * 8) return;
  int v = idx & 7;
  int lane = (idx >> 3) & 31;
  int rest = idx >> 8;
  int kk = rest % NKSTEP;
  int tile = rest / NKSTEP;
  int n = lane & 15, half = lane >> 4;
  int co = tile * 16 + n;
  int kcol = kk * 32 + 2 * v + half * 16;   // even, < 576
  int k = kcol >> 6, ci = kcol & 63;        // ci even
  uint32_t p0 = pack_split(w[(co * CCH + ci) * KSZ + k]);       // w shape (C,C,9)
  uint32_t p1 = pack_split(w[(co * CCH + ci + 1) * KSZ + k]);
  wH[idx] = (p0 & 0xffffu) | (p1 << 16);
  wL[idx] = (p0 >> 16) | (p1 & 0xffff0000u);
}

// ---------------- ragged segment pooling (deterministic, atomic-free) ----
// one block per bucket m; 64 threads = channels; t-range via binary search.
__global__ __launch_bounds__(64) void pool_kernel(
    const float* __restrict__ x, const float* __restrict__ w0,
    const float* __restrict__ b0, const float* __restrict__ weights,
    const float* __restrict__ pos, uint32_t* __restrict__ pooled) {
  int m = blockIdx.x;
  int c = threadIdx.x;
  float wk[KSZ];
#pragma unroll
  for (int k = 0; k < KSZ; ++k) wk[k] = w0[c * KSZ + k];
  float bc = b0[c];
  int t0, t1;
  {
    float key = (float)m;
    int lo = 0, hi = T_LEN;
    while (lo < hi) { int mid = (lo + hi) >> 1; if (pos[mid] < key) lo = mid + 1; else hi = mid; }
    t0 = lo;
  }
  if (m == M_LEN - 1) {
    t1 = T_LEN;                              // upper clip of buckets
  } else {
    float key = (float)(m + 1);
    int lo = 0, hi = T_LEN;
    while (lo < hi) { int mid = (lo + hi) >> 1; if (pos[mid] < key) lo = mid + 1; else hi = mid; }
    t1 = lo;
  }
  float num = 0.f, den = 0.f;
  for (int t = t0; t < t1; ++t) {
    float wt = weights[t];
    float s = bc;
#pragma unroll
    for (int k = 0; k < KSZ; ++k) {
      int u = t + k - KSZ / 2;
      float xv = (u >= 0 && u < T_LEN) ? x[u] : 0.f;
      s += wk[k] * xv;
    }
    num += wt * siluf_(s);
    den += wt;
  }
  float p = (den > 0.f) ? (num / den) : 0.f;
  pooled[(size_t)m * CCH + c] = pack_split(p);
}

// ---------------- stage 2 conv (K=9, 64->64) as split-bf16 WMMA GEMM -----
// one wave per 16-row M tile; 4 accumulators cover all 64 output channels.
// LDS holds hi/lo planes as bf16 PAIRS (adjacent channels share a u32), so
// A fragments are 8 direct ds_load_b32 and B fragments are 32B vector loads.
__global__ __launch_bounds__(256) void conv_gemm_kernel(
    const uint32_t* __restrict__ inp,    // [M,64] packed (hi|lo<<16) per element
    const uint32_t* __restrict__ wpkH,   // B fragment hi-plane pairs
    const uint32_t* __restrict__ wpkL,   // B fragment lo-plane pairs
    const float* __restrict__ bias,
    uint32_t* __restrict__ outP,         // packed output (layer 1)
    float* __restrict__ outF,            // f32 output (layer 2)
    int writePacked) {
  __shared__ uint32_t ldsH[8][24 * LDSPAIR];   // 24 rows x 32 channel-pairs (+pad)
  __shared__ uint32_t ldsL[8][24 * LDSPAIR];
  int wid = threadIdx.x >> 5, lane = threadIdx.x & 31;
  int mt = blockIdx.x * 8 + wid;
  bool active = (mt < MT);
  int m0 = mt << 4;
  if (active) {
    for (int i = lane; i < 24 * 32; i += 32) {   // rows m0-4 .. m0+19, pairwise
      int r = i >> 5, cp = i & 31;
      int g = m0 - 4 + r;
      uint32_t q0 = 0u, q1 = 0u;
      if (g >= 0 && g < M_LEN) {
        uint2 q = *(const uint2*)(inp + (size_t)g * CCH + 2 * cp);
        q0 = q.x; q1 = q.y;
      }
      // convert packed(elem) -> pair planes: lo16 of word = even channel
      ldsH[wid][r * LDSPAIR + cp] = (q0 & 0xffffu) | (q1 << 16);
      ldsL[wid][r * LDSPAIR + cp] = (q0 >> 16) | (q1 & 0xffff0000u);
    }
  }
  __syncthreads();
  if (!active) return;

  int half = lane >> 4, mrow = lane & 15;
  v8f acc[4];
  acc[0] = 0.f; acc[1] = 0.f; acc[2] = 0.f; acc[3] = 0.f;

  for (int kk = 0; kk < NKSTEP; ++kk) {
    // A fragment: element pairs (2v,2v+1) are adjacent K -> one u32 per VGPR
    v8u ah, al;
#pragma unroll
    for (int v = 0; v < 8; ++v) {
      int e = 2 * v;
      int Kloc = e + half * 16;
      int kcol = kk * 32 + Kloc;                 // even
      int k = kcol >> 6, cp = (kcol & 63) >> 1;
      int idx = (mrow + k) * LDSPAIR + cp;
      ah[v] = ldsH[wid][idx];
      al[v] = ldsL[wid][idx];
    }
    v16bf Ah = __builtin_bit_cast(v16bf, ah);
    v16bf Al = __builtin_bit_cast(v16bf, al);

#pragma unroll
    for (int t = 0; t < 4; ++t) {
      size_t b8 = ((size_t)((t * NKSTEP + kk) * 32 + lane)) << 3;
      v16bf Bh = __builtin_bit_cast(v16bf, *(const v8u*)(wpkH + b8));
      v16bf Bl = __builtin_bit_cast(v16bf, *(const v8u*)(wpkL + b8));
      if (kk + 1 < NKSTEP) {      // L2-resident B: prefetch next K-step
        size_t nb8 = ((size_t)((t * NKSTEP + kk + 1) * 32 + lane)) << 3;
        __builtin_prefetch(wpkH + nb8, 0, 1);
        __builtin_prefetch(wpkL + nb8, 0, 1);
      }
      // fp32 ~= (Ah+Al)(Bh+Bl): hi*hi + cross terms, f32 accumulate
      acc[t] = __builtin_amdgcn_wmma_f32_16x16x32_bf16(false, Ah, false, Bh, (short)0, acc[t], false, false);
      acc[t] = __builtin_amdgcn_wmma_f32_16x16x32_bf16(false, Ah, false, Bl, (short)0, acc[t], false, false);
      acc[t] = __builtin_amdgcn_wmma_f32_16x16x32_bf16(false, Al, false, Bh, (short)0, acc[t], false, false);
    }
  }

  // D layout: vgpr r -> row r + half*8, col = lane&15
#pragma unroll
  for (int t = 0; t < 4; ++t) {
    int co = t * 16 + mrow;
    float bsv = bias[co];
#pragma unroll
    for (int r = 0; r < 8; ++r) {
      int m = m0 + r + half * 8;
      float s = siluf_(acc[t][r] + bsv);
      size_t o = (size_t)m * CCH + co;
      if (writePacked) outP[o] = pack_split(s);
      else             outF[o] = s;
    }
  }
}

// -------------------------------------------------------------------------
extern "C" void kernel_launch(void* const* d_in, const int* in_sizes, int n_in,
                              void* d_out, int out_size, void* d_ws, size_t ws_size,
                              hipStream_t stream) {
  (void)in_sizes; (void)n_in; (void)out_size; (void)ws_size;
  const float* x  = (const float*)d_in[0];
  const float* w0 = (const float*)d_in[1];
  const float* b0 = (const float*)d_in[2];
  const float* wp = (const float*)d_in[3];
  const float* bp = (const float*)d_in[4];
  const float* w1 = (const float*)d_in[5];
  const float* b1 = (const float*)d_in[6];
  const float* w2 = (const float*)d_in[7];
  const float* b2 = (const float*)d_in[8];

  char* ws = (char*)d_ws;
  size_t off = 0;
  auto alloc = [&](size_t bytes) -> char* {
    char* p = ws + off;
    off += (bytes + 255) & ~(size_t)255;
    return p;
  };
  float*    weights = (float*)alloc((size_t)T_LEN * 4);
  float*    posbuf  = (float*)alloc((size_t)T_LEN * 4);   // moves, then cumsum in place
  float*    bsum    = (float*)alloc((size_t)NB * 4);
  uint32_t* wpkH1   = (uint32_t*)alloc((size_t)4 * NKSTEP * 32 * 8 * 4);
  uint32_t* wpkL1   = (uint32_t*)alloc((size_t)4 * NKSTEP * 32 * 8 * 4);
  uint32_t* wpkH2   = (uint32_t*)alloc((size_t)4 * NKSTEP * 32 * 8 * 4);
  uint32_t* wpkL2   = (uint32_t*)alloc((size_t)4 * NKSTEP * 32 * 8 * 4);
  uint32_t* pooled  = (uint32_t*)alloc((size_t)M_LEN * CCH * 4);
  uint32_t* h1      = (uint32_t*)alloc((size_t)M_LEN * CCH * 4);

  // stage 1: predictor + deterministic 3-phase scan
  predictor_kernel<<<NB, 256, 0, stream>>>(x, wp, bp, weights, posbuf, bsum);
  scan_bsums_kernel<<<1, 256, 0, stream>>>(bsum, NB);
  pos_scan_kernel<<<NB, 256, 0, stream>>>(posbuf, bsum);

  // weight prepack for both stage-2 layers (73 KB/plane; L2-resident)
  int wpk_threads = 4 * NKSTEP * 32 * 8;     // 18432
  pack_weights_kernel<<<(wpk_threads + 255) / 256, 256, 0, stream>>>(w1, wpkH1, wpkL1);
  pack_weights_kernel<<<(wpk_threads + 255) / 256, 256, 0, stream>>>(w2, wpkH2, wpkL2);

  // ragged pooling, conv0+SiLU fused on the fly
  pool_kernel<<<M_LEN, 64, 0, stream>>>(x, w0, b0, weights, posbuf, pooled);

  // stage 2: two 64->64 K=9 convs as split-bf16 WMMA GEMMs
  int gemm_blocks = (MT + 7) / 8;             // 2735
  conv_gemm_kernel<<<gemm_blocks, 256, 0, stream>>>(pooled, wpkH1, wpkL1, b1, h1, nullptr, 1);
  conv_gemm_kernel<<<gemm_blocks, 256, 0, stream>>>(h1, wpkH2, wpkL2, b2, nullptr, (float*)d_out, 0);
}
